// BaseController_55284819034692
// MI455X (gfx1250) — matrix operand order repaired and checked by reading, not compiled, not run on hardware
//
#include <hip/hip_runtime.h>
#include <hip/hip_bf16.h>
#include <stdint.h>

// ---------------------------------------------------------------------------
// Constants from the reference
// ---------------------------------------------------------------------------
#define NW     4096          // NUM_WORDS
#define HS     1024          // HSIZE
#define MAXW   20
#define EMBSZ  20
#define MLPH   1000
#define NT     38
#define NSPAN  (NW * MAXW)   // 81920
#define LOGN   (NSPAN * NT)  // 3112960
#define N2     2000          // concat [W1_start | W1_end] columns
#define NEGV   (-1e10f)

typedef __attribute__((ext_vector_type(16))) __bf16 bf16x16;
typedef __attribute__((ext_vector_type(8)))  float  f32x8;
typedef __attribute__((ext_vector_type(4)))  unsigned int u32x4;
typedef __attribute__((ext_vector_type(8)))  int  i32x8;
typedef __attribute__((ext_vector_type(4)))  int  i32x4;

union ABfrag { bf16x16 v; uint4 q[2]; };

// GEMM1 TDM tiling
#define BROWS 80                   // B-panel rows (N columns of GEMM) per block
#define KC    256                  // K-chunk elements staged per TDM transfer
#define ROWU  264                  // ushorts per padded LDS row (512B + 16B pad)
#define BUFU  (BROWS * ROWU)       // ushorts per LDS buffer (42240 B)

// ---------------------------------------------------------------------------
// Workspace layout (bytes). Total ~58 MB.
// ---------------------------------------------------------------------------
#define OFF_DOCB   ((size_t)0)                         // 4096*1024 bf16 = 8 MB
#define OFF_WT     (OFF_DOCB + (size_t)NW*HS*2)        // 2000*1024 bf16 = 4 MB
#define OFF_W2T    (OFF_WT   + (size_t)N2*HS*2)        // 48*1024 bf16
#define OFF_P      (OFF_W2T  + (size_t)48*1024*2)      // 4096*2000 f32 = 32.8 MB
#define OFF_WTERM  (OFF_P    + (size_t)NW*N2*4)        // 20*1024 f32
#define OFF_C0     (OFF_WTERM+ (size_t)MAXW*1024*4)    // 1024 f32
#define OFF_WS20   (OFF_C0   + (size_t)1024*4)         // 20 f32 (+pad)
#define OFF_LOGITS (OFF_WS20 + (size_t)256)            // 3112960 f32 = 12.5 MB
#define OFF_STATE  (OFF_LOGITS + (size_t)LOGN*4)       // a few uints
#define OFF_COUNTS (OFF_STATE + (size_t)256)           // 256 uints
#define OFF_TSC    (OFF_COUNTS+ (size_t)256*4)         // 2048 f32
#define OFF_TIX    (OFF_TSC  + (size_t)2048*4)         // 2048 int

// ---------------------------------------------------------------------------
// Helpers
// ---------------------------------------------------------------------------
__device__ __forceinline__ unsigned short f2bf(float x) {
  unsigned int u = __float_as_uint(x);
  unsigned int r = u + 0x7FFFu + ((u >> 16) & 1u);   // round-to-nearest-even
  return (unsigned short)(r >> 16);
}
__device__ __forceinline__ unsigned int fkey(float x) {   // monotonic f32->u32
  unsigned int u = __float_as_uint(x);
  unsigned int m = (unsigned int)((int)u >> 31);
  return u ^ (m | 0x80000000u);
}

// Issue one TDM transfer: 2D tile (KC cols x BROWS rows, bf16) from the
// row-major WT tensor (stride HS elems) into LDS at byte offset lds_off,
// with 16B padding appended per 512B row (pad_interval=128dw, pad_amount=4dw)
// so LDS row stride is 528B = 4-bank skew -> conflict-free ds_load_b128.
// Note: this toolchain exposes the 6-operand builtin
//   (u32x4 g0, i32x8 g1, i32x4 g2, i32x4 g3, i32x8 aux, i32 cpol)
__device__ __forceinline__ void tdm_load_panel(unsigned long long gaddr,
                                               unsigned int lds_off) {
  u32x4 g0;
  g0[0] = 1u;                                     // count=1, user descriptor
  g0[1] = lds_off;                                // lds_addr (bytes)
  g0[2] = (unsigned int)gaddr;                    // global_addr[31:0]
  g0[3] = (unsigned int)((gaddr >> 32) & 0x01FFFFFFull) | (2u << 30); // type=2
  i32x8 g1;
  g1[0] = (int)((1u << 16)      // data_size = 1 (2 bytes)
              | (1u << 20)      // pad_enable
              | (6u << 22)      // pad_interval code 6 = 128 DWORDs (512B)
              | (3u << 25));    // pad_amount  code 3 = 4 DWORDs (16B)
  g1[1] = (int)((unsigned)HS << 16);   // tensor_dim0 = 1024  (bits 63:48)
  g1[2] = (int)((unsigned)N2 << 16);   // tensor_dim1 = 2000  (bits 95:80)
  g1[3] = (int)((unsigned)KC << 16);   // tile_dim0   = 256   (bits 127:112)
  g1[4] = BROWS;                       // tile_dim1   = 80    (bits 143:128)
  g1[5] = HS;                          // tensor_dim0_stride = 1024 (bits 191:160)
  g1[6] = 0;
  g1[7] = 0;
  i32x4 z4 = {0, 0, 0, 0};
  i32x8 z8 = {0, 0, 0, 0, 0, 0, 0, 0};
  __builtin_amdgcn_tensor_load_to_lds(g0, g1, z4, z4, z8, 0);
}

// ---------------------------------------------------------------------------
// Packing kernels
// ---------------------------------------------------------------------------
__global__ void k_cvt_doc(const float* __restrict__ x, unsigned short* __restrict__ y, int n) {
  int i = blockIdx.x * 256 + threadIdx.x;
  if (i < n) y[i] = f2bf(x[i]);
}

// WT[n][k] (bf16, 2000x1024, K contiguous): n<1000 -> W1[k][n]; else W1[1024+k][n-1000]
__global__ void k_build_WT(const float* __restrict__ W1, unsigned short* __restrict__ WT) {
  int i = blockIdx.x * 256 + threadIdx.x;
  if (i >= N2 * HS) return;
  int n = i >> 10, kk = i & 1023;
  float v = (n < MLPH) ? W1[(size_t)kk * MLPH + n]
                       : W1[(size_t)(HS + kk) * MLPH + (n - MLPH)];
  WT[(size_t)n * HS + kk] = f2bf(v);
}

// W2T[n][k] (bf16, 48x1024): pad types 38->48, K 1000->1024 with zeros
__global__ void k_build_W2T(const float* __restrict__ W2, unsigned short* __restrict__ W2T) {
  int i = blockIdx.x * 256 + threadIdx.x;
  if (i >= 48 * 1024) return;
  int n = i >> 10, kk = i & 1023;
  float v = (n < NT && kk < MLPH) ? W2[(size_t)kk * NT + n] : 0.f;
  W2T[(size_t)n * 1024 + kk] = f2bf(v);
}

// widthTerm[w][j] = span_width_emb[w] . W1[2048+e][j];  c0[j] = b1[j] + dt_emb . W1[2068+e][j]
__global__ void k_wt_c0(const float* __restrict__ swe, const float* __restrict__ dte,
                        const float* __restrict__ W1, const float* __restrict__ b1,
                        const int* __restrict__ dtp,
                        float* __restrict__ wterm, float* __restrict__ c0) {
  int j = blockIdx.x * 256 + threadIdx.x;
  if (j >= 1024) return;
  if (j < MLPH) {
    for (int w = 0; w < MAXW; ++w) {
      float a = 0.f;
      for (int e = 0; e < EMBSZ; ++e) a += swe[w * EMBSZ + e] * W1[(size_t)(2 * HS + e) * MLPH + j];
      wterm[w * 1024 + j] = a;
    }
    int dt = *dtp;
    float c = b1[j];
    for (int e = 0; e < EMBSZ; ++e) c += dte[dt * EMBSZ + e] * W1[(size_t)(2 * HS + EMBSZ + e) * MLPH + j];
    c0[j] = c;
  } else {
    for (int w = 0; w < MAXW; ++w) wterm[w * 1024 + j] = 0.f;
    c0[j] = 0.f;
  }
}

// width prior scores: ws20[w] = relu(swp[w]@Wp1 + bp1) @ Wp2 + bp2
__global__ void k_wscore(const float* __restrict__ swp, const float* __restrict__ Wp1,
                         const float* __restrict__ bp1, const float* __restrict__ Wp2,
                         const float* __restrict__ bp2, float* __restrict__ ws20) {
  __shared__ float red[256];
  int w = blockIdx.x, t = threadIdx.x;
  float part = 0.f;
  for (int j = t; j < MLPH; j += 256) {
    float h = bp1[j];
    for (int e = 0; e < EMBSZ; ++e) h += swp[w * EMBSZ + e] * Wp1[(size_t)e * MLPH + j];
    h = h > 0.f ? h : 0.f;
    part += h * Wp2[j];
  }
  red[t] = part; __syncthreads();
  for (int s = 128; s > 0; s >>= 1) { if (t < s) red[t] += red[t + s]; __syncthreads(); }
  if (t == 0) ws20[w] = red[0] + bp2[0];
}

// ---------------------------------------------------------------------------
// GEMM1: P[4096x2000] = docB[4096x1024] @ WT^T  (bf16 WMMA, f32 accum)
// B panel (80 N-cols x 256 K-chunk) staged into LDS by the Tensor Data Mover,
// double-buffered, tracked with TENSORcnt. 4 waves/block share the panel;
// each wave computes one 16-row M tile x 80 cols. A rows stream from L2.
// ---------------------------------------------------------------------------
__global__ void __launch_bounds__(128) k_gemm1(const unsigned short* __restrict__ A,
                                               const unsigned short* __restrict__ B,
                                               float* __restrict__ P) {
  extern __shared__ unsigned short Bs[];          // 2 * BUFU ushorts (84.5 KB)
  const int lane = threadIdx.x & 31;
  const int wave = threadIdx.x >> 5;
  const int mtile = blockIdx.x * 4 + wave;        // 0..255
  const int nbase = blockIdx.y * BROWS;           // 0..1920
  const bool hi = lane >= 16;
  const int l16 = lane & 15;
  const unsigned short* arow = A + (size_t)(mtile * 16 + l16) * HS;
  const unsigned long long gpanel =
      (unsigned long long)(uintptr_t)(B + (size_t)nbase * HS);

  f32x8 acc[5] = {};

  if (wave == 0) {                                // prime the double buffer
    tdm_load_panel(gpanel + 0 * KC * 2, 0 * BUFU * 2);
    tdm_load_panel(gpanel + 1 * KC * 2, 1 * BUFU * 2);
  }

  for (int c = 0; c < HS / KC; ++c) {             // 4 K-chunks
    if (wave == 0) {
      if (c < HS / KC - 1) __builtin_amdgcn_s_wait_tensorcnt(1);  // oldest done
      else                 __builtin_amdgcn_s_wait_tensorcnt(0);
    }
    __syncthreads();
    const unsigned short* buf = Bs + (c & 1) * BUFU;
    const unsigned short* abase = arow + c * KC;
    for (int ks = 0; ks < KC; ks += 32) {
      ABfrag a;
      a.q[0] = *(const uint4*)(abase + ks + (hi ? 8 : 0));
      a.q[1] = *(const uint4*)(abase + ks + (hi ? 24 : 16));
#pragma unroll
      for (int t = 0; t < 5; ++t) {
        const unsigned short* bp = buf + (t * 16 + l16) * ROWU + ks + (hi ? 8 : 0);
        ABfrag b;
        b.q[0] = *(const uint4*)(bp);
        b.q[1] = *(const uint4*)(bp + 16);
        acc[t] = __builtin_amdgcn_wmma_f32_16x16x32_bf16(false, a.v, false, b.v,
                                                         (short)0, acc[t], false, false);
      }
    }
    __syncthreads();                              // all waves done with buf (c&1)
    if (wave == 0 && c + 2 < HS / KC)
      tdm_load_panel(gpanel + (size_t)(c + 2) * KC * 2, (c & 1) * BUFU * 2);
  }

#pragma unroll
  for (int t = 0; t < 5; ++t) {
    int ncol = nbase + t * 16 + l16;
    int mrow0 = mtile * 16 + (hi ? 8 : 0);
#pragma unroll
    for (int r = 0; r < 8; ++r) P[(size_t)(mrow0 + r) * N2 + ncol] = acc[t][r];
  }
}

// ---------------------------------------------------------------------------
// Fused span scoring: hidden tile (16 spans x 1024) in LDS (bf16), then
// WMMA vs W2T (3 N-tiles cover 48 padded types) -> logits (+prior, +mask).
// ---------------------------------------------------------------------------
__global__ void __launch_bounds__(128) k_span_logits(
    const float* __restrict__ P, const float* __restrict__ wterm,
    const float* __restrict__ c0, const float* __restrict__ ws20,
    const unsigned short* __restrict__ W2T, const float* __restrict__ b2,
    const int* __restrict__ sentmap, float* __restrict__ logits) {
  __shared__ __align__(16) unsigned short hid[16][1032];  // +8 pad -> 4-bank row skew
  const int tid = threadIdx.x;
  const int sb = blockIdx.x * 16;

  // Phase A: build hidden tile (coalesced reads of L2-resident P)
  for (int e = tid; e < 16 * 1024; e += 128) {
    int r = e >> 10, j = e & 1023;
    int span = sb + r;
    int word = span / MAXW, wdt = span - word * MAXW;
    int end = word + wdt; if (end > NW - 1) end = NW - 1;
    float v = 0.f;
    if (j < MLPH) {
      v = P[(size_t)word * N2 + j] + P[(size_t)end * N2 + MLPH + j]
        + wterm[wdt * 1024 + j] + c0[j];
      v = v > 0.f ? v : 0.f;
    }
    hid[r][j] = f2bf(v);
  }
  __syncthreads();

  // Phase B: 3 waves, one 16-type tile each; K-loop 32 over 1024
  const int wave = tid >> 5;
  if (wave < 3) {
    const int lane = tid & 31;
    const bool hi = lane >= 16;
    const int l16 = lane & 15;
    const int trow = wave * 16 + l16;
    f32x8 acc = {};
    for (int kk = 0; kk < 1024; kk += 32) {
      ABfrag a, b;
      a.q[0] = *(const uint4*)&hid[l16][kk + (hi ? 8 : 0)];
      a.q[1] = *(const uint4*)&hid[l16][kk + (hi ? 24 : 16)];
      const unsigned short* brow = W2T + (size_t)trow * 1024 + kk + (hi ? 8 : 0);
      b.q[0] = *(const uint4*)(brow);
      b.q[1] = *(const uint4*)(brow + 16);
      acc = __builtin_amdgcn_wmma_f32_16x16x32_bf16(false, a.v, false, b.v,
                                                    (short)0, acc, false, false);
    }
    if (trow < NT) {
#pragma unroll
      for (int r = 0; r < 8; ++r) {
        int row = r + (hi ? 8 : 0);
        int span = sb + row;
        int word = span / MAXW, wdt = span - word * MAXW;
        int endr = word + wdt;
        int end = endr > NW - 1 ? NW - 1 : endr;
        bool valid = (endr < NW) && (sentmap[word] == sentmap[end]);
        float v = valid ? (acc[r] + b2[trow] + ws20[wdt]) : NEGV;
        logits[(size_t)span * NT + trow] = v;
      }
    }
  }
}

// ---------------------------------------------------------------------------
// Top-k: 4-pass radix select + collect + bitonic sort
// state: [0]=prefix [1]=remaining [2]=count_above [3]=gt ctr [4]=eq ctr
// ---------------------------------------------------------------------------
__global__ void k_tk_init(unsigned int* counts, unsigned int* state, const int* kp) {
  int t = threadIdx.x;
  counts[t] = 0;
  if (t == 0) { state[0] = 0; state[1] = (unsigned int)*kp; state[2] = 0; state[3] = 0; state[4] = 0; }
}

__global__ void k_tk_hist(const float* __restrict__ lg, int n, int shift, unsigned int mask,
                          const unsigned int* __restrict__ state, unsigned int* __restrict__ counts) {
  __shared__ unsigned int h[256];
  int t = threadIdx.x;
  h[t] = 0; __syncthreads();
  unsigned int pref = state[0];
  for (int i = blockIdx.x * 256 + t; i < n; i += gridDim.x * 256) {
    unsigned int u = fkey(lg[i]);
    if ((u & mask) == pref) atomicAdd(&h[(u >> shift) & 255u], 1u);
  }
  __syncthreads();
  if (h[t]) atomicAdd(&counts[t], h[t]);
}

__global__ void k_tk_select(unsigned int* counts, unsigned int* state, int shift) {
  __shared__ unsigned int c[256];
  int t = threadIdx.x;
  c[t] = counts[t];
  counts[t] = 0;                      // reset for next pass
  __syncthreads();
  if (t == 0) {
    unsigned int rem = state[1], cum = 0, above = 0;
    int chosen = 0;
    for (int b = 255; b >= 0; --b) {
      unsigned int cc = c[b];
      if (cum + cc >= rem) { chosen = b; above = cum; break; }
      cum += cc;
    }
    state[0] |= (unsigned int)chosen << shift;
    state[1] = rem - above;
    state[2] += above;
  }
}

__global__ void k_tk_collect(const float* __restrict__ lg, int n, const int* __restrict__ kp,
                             unsigned int* __restrict__ state,
                             float* __restrict__ tsc, int* __restrict__ tix) {
  unsigned int T = state[0];
  unsigned int above = state[2];
  int k = *kp;
  for (int i = blockIdx.x * 256 + threadIdx.x; i < n; i += gridDim.x * 256) {
    float x = lg[i];
    unsigned int u = fkey(x);
    if (u > T) {
      unsigned int p = atomicAdd(&state[3], 1u);
      if (p < 2048u && (int)p < k) { tsc[p] = x; tix[p] = i; }
    } else if (u == T) {
      unsigned int q = atomicAdd(&state[4], 1u);
      unsigned int p = above + q;
      if (p < 2048u && (int)p < k) { tsc[p] = x; tix[p] = i; }
    }
  }
}

__global__ void __launch_bounds__(1024) k_tk_sort(float* __restrict__ tsc, int* __restrict__ tix,
                                                  const int* __restrict__ kp) {
  __shared__ float s[2048];
  __shared__ int   id[2048];
  int k = *kp;
  int tid = threadIdx.x;
  for (int i = tid; i < 2048; i += 1024) {
    if (i < k) { s[i] = tsc[i]; id[i] = tix[i]; }
    else       { s[i] = -3.4e38f; id[i] = 0x7FFFFFFF; }
  }
  __syncthreads();
  for (int size = 2; size <= 2048; size <<= 1) {
    for (int str = size >> 1; str > 0; str >>= 1) {
      for (int i = tid; i < 2048; i += 1024) {
        int j = i ^ str;
        if (j > i) {
          float si = s[i], sj = s[j];
          int ii = id[i], ij = id[j];
          bool pre = (si > sj) || (si == sj && ii < ij);  // descending, tie: low idx
          bool up = ((i & size) == 0);
          if (up ? !pre : pre) { s[i] = sj; s[j] = si; id[i] = ij; id[j] = ii; }
        }
      }
      __syncthreads();
    }
  }
  for (int i = tid; i < 2048; i += 1024) {
    if (i < k) { tsc[i] = s[i]; tix[i] = id[i]; }
  }
}

__global__ void k_zero_mask(float* __restrict__ out, const int* __restrict__ kp, int n) {
  int k = *kp;
  int i = blockIdx.x * 256 + threadIdx.x;
  if (i < n) out[(size_t)4 * k + i] = 0.f;
}

__global__ void k_final(const float* __restrict__ tsc, const int* __restrict__ tix,
                        const int* __restrict__ kp, float* __restrict__ out) {
  int k = *kp;
  int i = blockIdx.x * 256 + threadIdx.x;
  if (i >= k) return;
  int flat = tix[i];
  int cand = flat / NT, t = flat - cand * NT;
  int word = cand / MAXW, wdt = cand - word * MAXW;
  int endr = word + wdt;
  int end = endr > NW - 1 ? NW - 1 : endr;
  out[i]         = (float)word;      // topk_starts
  out[k + i]     = (float)end;       // topk_ends
  out[2 * k + i] = tsc[i];           // topk_scores
  out[3 * k + i] = (float)t;         // topk_ment_type
  out[(size_t)4 * k + cand] = 1.0f;  // topk_mask (reference's cand-index scatter)
}

// ---------------------------------------------------------------------------
extern "C" void kernel_launch(void* const* d_in, const int* in_sizes, int n_in,
                              void* d_out, int out_size, void* d_ws, size_t ws_size,
                              hipStream_t stream) {
  const float* encoded = (const float*)d_in[0];
  const int*   sentmap = (const int*)d_in[1];
  const float* swe     = (const float*)d_in[2];
  const float* swp     = (const float*)d_in[3];
  const float* dte     = (const float*)d_in[4];
  const float* W1      = (const float*)d_in[5];
  const float* b1      = (const float*)d_in[6];
  const float* W2      = (const float*)d_in[7];
  const float* b2      = (const float*)d_in[8];
  const float* Wp1     = (const float*)d_in[9];
  const float* bp1     = (const float*)d_in[10];
  const float* Wp2     = (const float*)d_in[11];
  const float* bp2     = (const float*)d_in[12];
  const int*   dtp     = (const int*)d_in[13];
  const int*   kp      = (const int*)d_in[14];
  float* out = (float*)d_out;

  uint8_t* w = (uint8_t*)d_ws;   // requires ~58 MB of workspace
  unsigned short* docB  = (unsigned short*)(w + OFF_DOCB);
  unsigned short* WT    = (unsigned short*)(w + OFF_WT);
  unsigned short* W2T   = (unsigned short*)(w + OFF_W2T);
  float*          P     = (float*)(w + OFF_P);
  float*          wterm = (float*)(w + OFF_WTERM);
  float*          c0    = (float*)(w + OFF_C0);
  float*          ws20  = (float*)(w + OFF_WS20);
  float*          lg    = (float*)(w + OFF_LOGITS);
  unsigned int*   state = (unsigned int*)(w + OFF_STATE);
  unsigned int*   cnts  = (unsigned int*)(w + OFF_COUNTS);
  float*          tsc   = (float*)(w + OFF_TSC);
  int*            tix   = (int*)(w + OFF_TIX);

  int khost = (out_size - LOGN) / 4;
  if (khost < 1) khost = 1;

  // Packing / small precompute
  k_cvt_doc  <<<(NW * HS + 255) / 256, 256, 0, stream>>>(encoded, docB, NW * HS);
  k_build_WT <<<(N2 * HS + 255) / 256, 256, 0, stream>>>(W1, WT);
  k_build_W2T<<<(48 * 1024 + 255) / 256, 256, 0, stream>>>(W2, W2T);
  k_wt_c0    <<<4, 256, 0, stream>>>(swe, dte, W1, b1, dtp, wterm, c0);
  k_wscore   <<<MAXW, 256, 0, stream>>>(swp, Wp1, bp1, Wp2, bp2, ws20);

  // Stage 1: doc @ [W1s|W1e]  (bf16 WMMA, TDM-staged B panels, double-buffered)
  k_gemm1<<<dim3(64, 25), 128, 2 * BUFU * 2, stream>>>(docB, WT, P);

  // Stage 2: fused span hidden + logits (bf16 WMMA via LDS tiles)
  k_span_logits<<<NSPAN / 16, 128, 0, stream>>>(P, wterm, c0, ws20, W2T, b2, sentmap, lg);

  // Top-k radix select
  k_tk_init<<<1, 256, 0, stream>>>(cnts, state, kp);
  const int shifts[4] = {24, 16, 8, 0};
  for (int p = 0; p < 4; ++p) {
    unsigned int mask = (p == 0) ? 0u : (0xFFFFFFFFu << (shifts[p] + 8));
    k_tk_hist  <<<2048, 256, 0, stream>>>(lg, LOGN, shifts[p], mask, state, cnts);
    k_tk_select<<<1, 256, 0, stream>>>(cnts, state, shifts[p]);
  }
  k_tk_collect<<<2048, 256, 0, stream>>>(lg, LOGN, kp, state, tsc, tix);
  k_tk_sort<<<1, 1024, 0, stream>>>(tsc, tix, kp);

  // Outputs
  k_zero_mask<<<(LOGN + 255) / 256, 256, 0, stream>>>(out, kp, LOGN);
  k_final<<<(khost + 255) / 256, 256, 0, stream>>>(tsc, tix, kp, out);
}
